// LigRecDynamicsGVP_11948599017847
// MI455X (gfx1250) — matrix operand order repaired and checked by reading, not compiled, not run on hardware
//
#include <hip/hip_runtime.h>
#include <cstdint>
#include <cstddef>

typedef _Float16 h16;
typedef __attribute__((ext_vector_type(16))) _Float16 v16h;
typedef __attribute__((ext_vector_type(8)))  _Float16 v8h;
typedef __attribute__((ext_vector_type(8)))  float    v8f;

namespace {

constexpr int kB    = 64;
constexpr int kNL   = 256;
constexpr int kNK   = 20;
constexpr int kSL   = 16;
constexpr int kSK   = 256;
constexpr int kHID  = 128;
constexpr int kVEC  = 16;
constexpr int kLLK  = 16;
constexpr int kKLK  = 8;
constexpr int kNCONV= 4;
constexpr int ELDA  = 160;             // padded K (edge scalar rows): 146/144 -> 160
constexpr int NLDA  = 288;             // padded K (node scalar rows): 288 / 257 -> 288
constexpr int kNN   = kB * kNL;        // 16384 ligand nodes
constexpr int kNKP  = kB * kNK;        // 1280 kp nodes

__device__ __forceinline__ float fast_sigmoid(float x) {
  return __builtin_amdgcn_rcpf(1.f + __expf(-x));
}
__device__ __forceinline__ float act_apply(float v, int act) {
  if (act == 1) return v * fast_sigmoid(v);  // silu
  if (act == 2) return fast_sigmoid(v);      // sigmoid
  return v;
}

// ---------------- WMMA GEMM (templated on padded K, fully unrolled) -----------------------
// One wave -> 64x16 output strip (4 M-tiles share each B fragment).
// B is pre-packed in WMMA fragment layout: element ((tn*NKC + kc)*32 + lane)*16 + h.
// If gateOut != null: result is a gate; writes gate * Vu (3 components per element).
template <int KPAD>
__global__ void k_gemm_t(const h16* __restrict__ A, int lda,
                         const h16* __restrict__ Bw, int N,
                         const float* __restrict__ bias,
                         int M, int Mtiles4, int act,
                         float* __restrict__ outF, int ldoF,
                         h16* __restrict__ outH, int ldoH,
                         const float* __restrict__ gateVu,
                         float* __restrict__ gateOut)
{
  constexpr int NKC = KPAD / 32;
  const int wave = (blockIdx.x * blockDim.x + threadIdx.x) >> 5;
  const int lane = threadIdx.x & 31;
  const int ntN  = N >> 4;
  const int tm   = wave / ntN;
  const int tn   = wave - tm * ntN;
  if (tm >= Mtiles4) return;

  v8f acc[4];
#pragma unroll
  for (int mt = 0; mt < 4; ++mt) acc[mt] = (v8f){0.f,0.f,0.f,0.f,0.f,0.f,0.f,0.f};

  const int kbA = (lane >> 4) ? 8 : 0;
  const h16*  Ap0 = A + (size_t)(tm * 64 + (lane & 15)) * lda;
  const v16h* Bp  = (const v16h*)Bw + (size_t)tn * NKC * 32 + lane;

#pragma unroll
  for (int kc = 0; kc < NKC; ++kc) {
    const v16h bf = Bp[kc * 32];
#pragma unroll
    for (int mt = 0; mt < 4; ++mt) {
      const h16* Ap = Ap0 + (size_t)mt * 16 * lda + kc * 32 + kbA;
      v8h alo = *(const v8h*)Ap;
      v8h ahi = *(const v8h*)(Ap + 16);
      v16h af = __builtin_shufflevector(alo, ahi,
                                        0,1,2,3,4,5,6,7,8,9,10,11,12,13,14,15);
      acc[mt] = __builtin_amdgcn_wmma_f32_16x16x32_f16(false, af, false, bf,
                                                       (short)0, acc[mt], false, false);
    }
  }

  const int col  = tn * 16 + (lane & 15);
  const float bv = bias ? bias[col] : 0.f;
#pragma unroll
  for (int mt = 0; mt < 4; ++mt) {
#pragma unroll
    for (int r = 0; r < 8; ++r) {
      const int row = tm * 64 + mt * 16 + r + ((lane >> 4) << 3);
      if (row >= M) continue;
      float v = act_apply(acc[mt][r] + bv, act);
      if (gateOut) {
        const size_t p = ((size_t)row * N + col) * 3;
        gateOut[p]   = v * gateVu[p];
        gateOut[p+1] = v * gateVu[p+1];
        gateOut[p+2] = v * gateVu[p+2];
      } else {
        if (outF) outF[(size_t)row * ldoF + col] = v;
        if (outH) outH[(size_t)row * ldoH + col] = (h16)v;
      }
    }
  }
}

// ---------------- f32 weights -> f16, packed into WMMA B-fragment layout ------------------
__global__ void k_cvt_pack(const float* __restrict__ W, int K, int N, int Kpad,
                           h16* __restrict__ dst)
{
  int t = blockIdx.x * blockDim.x + threadIdx.x;
  int total = Kpad * N;
  if (t >= total) return;
  int h     = t & 15;
  int lane  = (t >> 4) & 31;
  int chunk = t >> 9;                 // tn*NKC + kc
  int nkc   = Kpad >> 5;
  int tn    = chunk / nkc, kc = chunk - tn * nkc;
  int n = tn * 16 + (lane & 15);
  int k = kc * 32 + ((lane >> 4) << 3) + (h & 7) + ((h >> 3) << 4);
  dst[t] = (h16)((k < K) ? W[(size_t)k * N + n] : 0.f);
}

// ---------------- brute-force KNN (top-K smallest squared distance, sorted) ---------------
__global__ void k_knn(const float* __restrict__ xq, const float* __restrict__ xc,
                      int nq, int nc, int K, int excl, int* __restrict__ idx)
{
  int q = blockIdx.x * blockDim.x + threadIdx.x;
  int total = kB * nq;
  if (q >= total) return;
  int b = q / nq, i = q - b * nq;
  const float* xp = xq + ((size_t)b * nq + i) * 3;
  float qx = xp[0], qy = xp[1], qz = xp[2];
  const float* xb = xc + (size_t)b * nc * 3;
  float bd[16]; int bi[16];
  for (int k = 0; k < K; ++k) { bd[k] = 3.0e38f; bi[k] = 0; }
  for (int j = 0; j < nc; ++j) {
    float dx = xb[j*3]-qx, dy = xb[j*3+1]-qy, dz = xb[j*3+2]-qz;
    float d = dx*dx + dy*dy + dz*dz;
    if (excl && j == i) d += 1e9f;
    if (d < bd[K-1]) {
      int p = K - 1;
      while (p > 0 && bd[p-1] > d) { bd[p] = bd[p-1]; bi[p] = bi[p-1]; --p; }
      bd[p] = d; bi[p] = j;
    }
  }
  for (int k = 0; k < K; ++k) idx[(size_t)q * K + k] = bi[k];
}

// ---------------- encoder A-row builder: [h0, t, 0-pad] -> f16 ----------------------------
__global__ void k_build_enc(const float* __restrict__ h0, const float* __restrict__ ts,
                            int nper, int S, int Kpad, h16* __restrict__ A, int lda, int M)
{
  int m = blockIdx.x * blockDim.x + threadIdx.x;
  if (m >= M) return;
  int b = m / nper;
  const float* src = h0 + (size_t)m * S;
  h16* Ar = A + (size_t)m * lda;
  for (int c = 0; c < S; ++c) Ar[c] = (h16)src[c];
  Ar[S] = (h16)ts[b];
  for (int c = S + 1; c < Kpad; ++c) Ar[c] = (h16)0.f;
}

// ---------------- layernorm (optionally with residual), row = 128 -------------------------
__global__ void k_ln(const float* __restrict__ x, const float* __restrict__ res,
                     const float* __restrict__ g, const float* __restrict__ bb,
                     float* __restrict__ out, int M)
{
  int m = blockIdx.x * blockDim.x + threadIdx.x;
  if (m >= M) return;
  const float* xp = x + (size_t)m * kHID;
  const float* rp = res ? res + (size_t)m * kHID : nullptr;
  float mu = 0.f;
  for (int c = 0; c < kHID; ++c) mu += xp[c] + (rp ? rp[c] : 0.f);
  mu *= (1.f / kHID);
  float var = 0.f;
  for (int c = 0; c < kHID; ++c) { float t = xp[c] + (rp ? rp[c] : 0.f) - mu; var += t * t; }
  var *= (1.f / kHID);
  float inv = rsqrtf(var + 1e-5f);
  float* op = out + (size_t)m * kHID;
  for (int c = 0; c < kHID; ++c) {
    float t = xp[c] + (rp ? rp[c] : 0.f);
    op[c] = (t - mu) * inv * g[c] + bb[c];
  }
}

__global__ void k_zero(float* __restrict__ a, int n) {
  int t = blockIdx.x * blockDim.x + threadIdx.x;
  if (t < n) a[t] = 0.f;
}
__global__ void k_add(float* __restrict__ a, const float* __restrict__ b, int n) {
  int t = blockIdx.x * blockDim.x + threadIdx.x;
  if (t < n) a[t] += b[t];
}
__global__ void k_castrow(const float* __restrict__ src, h16* __restrict__ dst,
                          int ldd, int cols, int M)
{
  int t = blockIdx.x * blockDim.x + threadIdx.x;
  int total = M * cols;
  if (t >= total) return;
  int m = t / cols, c = t - m * cols;
  dst[(size_t)m * ldd + c] = (h16)src[t];
}

// ---------------- edge builders ------------------------------------------------------------
__global__ void k_build_ll(const float* __restrict__ s_l, const float* __restrict__ v_l,
                           const float* __restrict__ x_l, const int* __restrict__ ll,
                           int node0, int nNodes, h16* __restrict__ A, float* __restrict__ V17)
{
  int e = blockIdx.x * blockDim.x + threadIdx.x;
  int E = nNodes * kLLK;
  if (e >= E) return;
  int nl = e / kLLK, k = e - nl * kLLK;
  int g = node0 + nl;
  int b = g / kNL, i = g - b * kNL;
  int j = ll[(size_t)g * kLLK + k];
  size_t src = (size_t)b * kNL + j;
  size_t dst = (size_t)b * kNL + i;
  const float* ss = s_l + src * kHID;
  h16* Ar = A + (size_t)e * ELDA;
  for (int c = 0; c < kHID; ++c) Ar[c] = (h16)ss[c];
  float dx = x_l[src*3]   - x_l[dst*3];
  float dy = x_l[src*3+1] - x_l[dst*3+1];
  float dz = x_l[src*3+2] - x_l[dst*3+2];
  Ar[kHID] = (h16)sqrtf(dx*dx + dy*dy + dz*dz + 1e-8f);
  const float* vv = v_l + src * kVEC * 3;
  float* Vo = V17 + (size_t)e * 51;
  for (int u = 0; u < kVEC * 3; ++u) Vo[u] = vv[u];
  Vo[48] = dx; Vo[49] = dy; Vo[50] = dz;
}

__global__ void k_build_kl(const float* __restrict__ s_k, const float* __restrict__ v_k,
                           const float* __restrict__ x_k, const float* __restrict__ x_l,
                           const int* __restrict__ kl, int kp0, int nKp,
                           h16* __restrict__ A, float* __restrict__ V17)
{
  int e = blockIdx.x * blockDim.x + threadIdx.x;
  int E = nKp * kKLK;
  if (e >= E) return;
  int kl_ = e / kKLK, m = e - kl_ * kKLK;
  int g = kp0 + kl_;
  int b = g / kNK;
  int j = kl[(size_t)g * kKLK + m];
  const float* ss = s_k + (size_t)g * kHID;
  h16* Ar = A + (size_t)e * ELDA;
  for (int c = 0; c < kHID; ++c) Ar[c] = (h16)ss[c];
  const float* xs = x_k + (size_t)g * 3;
  const float* xd = x_l + ((size_t)b * kNL + j) * 3;
  float dx = xs[0]-xd[0], dy = xs[1]-xd[1], dz = xs[2]-xd[2];
  Ar[kHID] = (h16)sqrtf(dx*dx + dy*dy + dz*dz + 1e-8f);
  const float* vv = v_k + (size_t)g * kVEC * 3;
  float* Vo = V17 + (size_t)e * 51;
  for (int u = 0; u < kVEC * 3; ++u) Vo[u] = vv[u];
  Vo[48] = dx; Vo[49] = dy; Vo[50] = dz;
}

// ---------------- GVP vector path: Vh = Wh^T v, sh = |Vh|, Vu = Wu^T Vh -------------------
template <int VI, int H, int VO>
__global__ void k_gvp_vec(const float* __restrict__ Vin, const float* __restrict__ Wh,
                          const float* __restrict__ Wu, int M,
                          h16* __restrict__ Ash, int lda, int shOff, int Kpad,
                          float* __restrict__ Vu)
{
  int m = blockIdx.x * blockDim.x + threadIdx.x;
  if (m >= M) return;
  float v[VI][3];
  const float* vp = Vin + (size_t)m * VI * 3;
#pragma unroll
  for (int i = 0; i < VI; ++i) { v[i][0]=vp[i*3]; v[i][1]=vp[i*3+1]; v[i][2]=vp[i*3+2]; }
  float vh[H][3];
  h16* Ar = Ash + (size_t)m * lda;
  for (int hh = 0; hh < H; ++hh) {
    float a0=0.f, a1=0.f, a2=0.f;
#pragma unroll
    for (int i = 0; i < VI; ++i) {
      float w = Wh[i * H + hh];
      a0 += w * v[i][0]; a1 += w * v[i][1]; a2 += w * v[i][2];
    }
    vh[hh][0]=a0; vh[hh][1]=a1; vh[hh][2]=a2;
    Ar[shOff + hh] = (h16)sqrtf(a0*a0 + a1*a1 + a2*a2 + 1e-8f);
  }
  for (int c = shOff + H; c < Kpad; ++c) Ar[c] = (h16)0.f;
  float* up = Vu + (size_t)m * VO * 3;
  for (int u = 0; u < VO; ++u) {
    float a0=0.f, a1=0.f, a2=0.f;
#pragma unroll
    for (int hh = 0; hh < H; ++hh) {
      float w = Wu[hh * VO + u];
      a0 += w * vh[hh][0]; a1 += w * vh[hh][1]; a2 += w * vh[hh][2];
    }
    up[u*3]=a0; up[u*3+1]=a1; up[u*3+2]=a2;
  }
}

// ---------------- aggregation -------------------------------------------------------------
__global__ void k_agg_ll_s(const h16* __restrict__ ms, int node0, int nNodes,
                           float* __restrict__ aggs)
{
  int t = blockIdx.x * blockDim.x + threadIdx.x;
  int total = nNodes * kHID;
  if (t >= total) return;
  int nl = t / kHID, c = t - nl * kHID;
  float a = 0.f;
  const h16* base = ms + (size_t)nl * kLLK * ELDA + c;
  for (int k = 0; k < kLLK; ++k) a += (float)base[(size_t)k * ELDA];
  aggs[(size_t)(node0 + nl) * kHID + c] = a;
}
__global__ void k_agg_ll_v(const float* __restrict__ mv, int node0, int nNodes,
                           float* __restrict__ aggv)
{
  int t = blockIdx.x * blockDim.x + threadIdx.x;
  int total = nNodes * 48;
  if (t >= total) return;
  int nl = t / 48, r = t - nl * 48;
  float a = 0.f;
  for (int k = 0; k < kLLK; ++k) a += mv[(size_t)(nl * kLLK + k) * 48 + r];
  aggv[(size_t)(node0 + nl) * 48 + r] = a;
}
__global__ void k_scat_s(const h16* __restrict__ ms, const int* __restrict__ kl,
                         int kp0, int nKp, float* __restrict__ aggs)
{
  int t = blockIdx.x * blockDim.x + threadIdx.x;
  int total = nKp * kKLK * kHID;
  if (t >= total) return;
  int e = t / kHID, c = t - e * kHID;
  int kl_ = e / kKLK, m = e - kl_ * kKLK;
  int g = kp0 + kl_;
  int b = g / kNK;
  int j = kl[(size_t)g * kKLK + m];
  atomicAdd(&aggs[((size_t)b * kNL + j) * kHID + c], (float)ms[(size_t)e * ELDA + c]);
}
__global__ void k_scat_v(const float* __restrict__ mv, const int* __restrict__ kl,
                         int kp0, int nKp, float* __restrict__ aggv)
{
  int t = blockIdx.x * blockDim.x + threadIdx.x;
  int total = nKp * kKLK * 48;
  if (t >= total) return;
  int e = t / 48, r = t - e * 48;
  int kl_ = e / kKLK, m = e - kl_ * kKLK;
  int g = kp0 + kl_;
  int b = g / kNK;
  int j = kl[(size_t)g * kKLK + m];
  atomicAdd(&aggv[((size_t)b * kNL + j) * 48 + r], mv[(size_t)e * 48 + r]);
}

// ---------------- update-chain input builder ----------------------------------------------
__global__ void k_build_upd(const float* __restrict__ s_l, const float* __restrict__ aggs,
                            const float* __restrict__ v_l, const float* __restrict__ aggv,
                            h16* __restrict__ A, float* __restrict__ V32)
{
  int n = blockIdx.x * blockDim.x + threadIdx.x;
  if (n >= kNN) return;
  h16* Ar = A + (size_t)n * NLDA;
  for (int c = 0; c < kHID; ++c) {
    Ar[c]        = (h16)s_l[(size_t)n * kHID + c];
    Ar[kHID + c] = (h16)aggs[(size_t)n * kHID + c];
  }
  float* Vo = V32 + (size_t)n * 96;
  for (int r = 0; r < 48; ++r) {
    Vo[r]      = v_l[(size_t)n * 48 + r];
    Vo[48 + r] = aggv[(size_t)n * 48 + r];
  }
}

// ---------------- last noise GVP: gate (no sigmoid, v_out=1) ------------------------------
__global__ void k_gate1(const float* __restrict__ sfeat, const float* __restrict__ Wg,
                        const float* __restrict__ bg, const float* __restrict__ Vu,
                        float* __restrict__ outv, int M)
{
  int m = blockIdx.x * blockDim.x + threadIdx.x;
  if (m >= M) return;
  float acc = bg[0];
  const float* sp = sfeat + (size_t)m * 64;
  for (int k = 0; k < 64; ++k) acc += sp[k] * Wg[k];
  const float* vp = Vu + (size_t)m * 3;
  float* o = outv + (size_t)m * 3;
  o[0] = acc * vp[0]; o[1] = acc * vp[1]; o[2] = acc * vp[2];
}

// ---------------- host helpers -------------------------------------------------------------
struct GvpRef { const float *Wh, *Wu, *bg, *bs; const h16 *Ws, *Wg; };

inline void launch_gemm(hipStream_t s, const h16* A, int lda, const h16* Bw, int N,
                        const float* bias, int M, int Kpad, int act,
                        float* outF, int ldoF, h16* outH, int ldoH,
                        const float* gateVu, float* gateOut)
{
  int Mtiles4 = M >> 6;                 // all M here are multiples of 64
  int waves   = Mtiles4 * (N >> 4);
  int blocks  = (waves + 7) / 8;
#define GEMM_CASE(KP) \
  k_gemm_t<KP><<<blocks, 256, 0, s>>>(A, lda, Bw, N, bias, M, Mtiles4, act, \
                                      outF, ldoF, outH, ldoH, gateVu, gateOut)
  switch (Kpad) {
    case 32:  GEMM_CASE(32);  break;
    case 64:  GEMM_CASE(64);  break;
    case 128: GEMM_CASE(128); break;
    case 160: GEMM_CASE(160); break;
    case 288: GEMM_CASE(288); break;
    default: break;
  }
#undef GEMM_CASE
}
inline void launch_cvt(hipStream_t s, const float* W, int K, int N, int Kpad, h16* dst) {
  int total = Kpad * N;
  k_cvt_pack<<<(total + 255) / 256, 256, 0, s>>>(W, K, N, Kpad, dst);
}

// three-GVP message chain over M edge rows (A ping-pong, Vu/Vg reuse)
void run_msg_chain(hipStream_t s, int M, h16* Aping, h16* Apong,
                   float* V17, float* Vu, float* Vg,
                   const GvpRef& g1, const GvpRef& g2, const GvpRef& g3)
{
  int gE = (M + 255) / 256;
  k_gvp_vec<17,17,16><<<gE,256,0,s>>>(V17, g1.Wh, g1.Wu, M, Aping, ELDA, 129, ELDA, Vu);
  launch_gemm(s, Aping, ELDA, g1.Ws, kHID, g1.bs, M, ELDA, 1, nullptr,0, Apong, ELDA, nullptr, nullptr);
  launch_gemm(s, Apong, ELDA, g1.Wg, kVEC, g1.bg, M, kHID, 2, nullptr,0, nullptr,0, Vu, Vg);
  k_gvp_vec<16,16,16><<<gE,256,0,s>>>(Vg, g2.Wh, g2.Wu, M, Apong, ELDA, 128, ELDA, Vu);
  launch_gemm(s, Apong, ELDA, g2.Ws, kHID, g2.bs, M, ELDA, 1, nullptr,0, Aping, ELDA, nullptr, nullptr);
  launch_gemm(s, Aping, ELDA, g2.Wg, kVEC, g2.bg, M, kHID, 2, nullptr,0, nullptr,0, Vu, Vg);
  k_gvp_vec<16,16,16><<<gE,256,0,s>>>(Vg, g3.Wh, g3.Wu, M, Aping, ELDA, 128, ELDA, Vu);
  launch_gemm(s, Aping, ELDA, g3.Ws, kHID, g3.bs, M, ELDA, 1, nullptr,0, Apong, ELDA, nullptr, nullptr);
  launch_gemm(s, Apong, ELDA, g3.Wg, kVEC, g3.bg, M, kHID, 2, nullptr,0, nullptr,0, Vu, Vg);
}

struct Bump {
  char* base; size_t off, cap; bool ok;
  void* get(size_t bytes) {
    size_t o = (off + 255) & ~(size_t)255;
    off = o + bytes;
    if (off > cap) ok = false;
    return base + o;
  }
};

} // namespace

extern "C" void kernel_launch(void* const* d_in, const int* in_sizes, int n_in,
                              void* d_out, int out_size, void* d_ws, size_t ws_size,
                              hipStream_t stream)
{
  (void)in_sizes; (void)out_size;
  // 6 tensor inputs + 228 param leaves (jax tree-flatten order: sorted dict keys)
  if (n_in < 234) return;

  const float* lig_h0 = (const float*)d_in[0];
  const float* kp_h0  = (const float*)d_in[1];
  const float* lig_x  = (const float*)d_in[2];
  const float* kp_x   = (const float*)d_in[3];
  const float* kp_v   = (const float*)d_in[4];
  const float* tstep  = (const float*)d_in[5];
  auto F = [&](int i) { return (const float*)d_in[i]; };

  // param leaf index map: convs first, per-conv 50 leaves
  const int KP_W = 206, KP_B = 207, KP_BETA = 208, KP_G = 209;
  const int LIG_W = 210, LIG_B = 211, LIG_BETA = 212, LIG_G = 213;
  const int NOISE0 = 214, OUT_W = 232, OUT_B = 233;

  // ---- workspace carve (adaptive edge chunking to fit ws_size) ----
  h16 *Aping=nullptr,*Apong=nullptr,*nA1=nullptr,*nA2=nullptr,*s3h=nullptr;
  h16 *wS[3]={nullptr,nullptr,nullptr},*wG[3]={nullptr,nullptr,nullptr},*wEnc=nullptr,*wOut=nullptr;
  float *V17=nullptr,*Vu=nullptr,*Vg=nullptr,*nV32=nullptr,*nVu=nullptr,*nVg=nullptr;
  float *s_l=nullptr,*s_k=nullptr,*v_l=nullptr,*aggs=nullptr,*aggv=nullptr,*s_tmp=nullptr;
  int *llidx=nullptr,*klidx=nullptr;
  int chunkNodes = 0;
  const int cands[3] = {4096, 1024, 256};
  for (int ci = 0; ci < 3; ++ci) {
    int cn = cands[ci];
    size_t CE = (size_t)cn * kLLK;
    Bump bp{(char*)d_ws, 0, ws_size, true};
    Aping = (h16*)bp.get(CE * ELDA * sizeof(h16));
    Apong = (h16*)bp.get(CE * ELDA * sizeof(h16));
    V17   = (float*)bp.get(CE * 51 * sizeof(float));
    Vu    = (float*)bp.get(CE * 48 * sizeof(float));
    Vg    = (float*)bp.get(CE * 48 * sizeof(float));
    nA1   = (h16*)bp.get((size_t)kNN * NLDA * sizeof(h16));
    nA2   = (h16*)bp.get((size_t)kNN * NLDA * sizeof(h16));
    nV32  = (float*)bp.get((size_t)kNN * 96 * sizeof(float));
    nVu   = (float*)bp.get((size_t)kNN * 48 * sizeof(float));
    nVg   = (float*)bp.get((size_t)kNN * 48 * sizeof(float));
    s_l   = (float*)bp.get((size_t)kNN * kHID * sizeof(float));
    s_k   = (float*)bp.get((size_t)kNKP * kHID * sizeof(float));
    v_l   = (float*)bp.get((size_t)kNN * 48 * sizeof(float));
    aggs  = (float*)bp.get((size_t)kNN * kHID * sizeof(float));
    aggv  = (float*)bp.get((size_t)kNN * 48 * sizeof(float));
    s_tmp = (float*)bp.get((size_t)kNN * kHID * sizeof(float));
    s3h   = (h16*)bp.get((size_t)kNN * 64 * sizeof(h16));
    llidx = (int*)bp.get((size_t)kNN * kLLK * sizeof(int));
    klidx = (int*)bp.get((size_t)kNKP * kKLK * sizeof(int));
    for (int i = 0; i < 3; ++i) {
      wS[i] = (h16*)bp.get((size_t)NLDA * kHID * sizeof(h16));
      wG[i] = (h16*)bp.get((size_t)kHID * kVEC * sizeof(h16));
    }
    wEnc = (h16*)bp.get((size_t)NLDA * kHID * sizeof(h16));
    wOut = (h16*)bp.get((size_t)64 * 16 * sizeof(h16));
    if (bp.ok) { chunkNodes = cn; break; }
  }
  if (!chunkNodes) return;

  auto mkref = [&](int gb, const h16* Ws, const h16* Wg) -> GvpRef {
    return GvpRef{ F(gb + 1), F(gb + 3), F(gb + 4), F(gb + 5), Ws, Wg };
  };

  const int gN  = (kNN + 255) / 256;
  const int gKP = (kNKP + 255) / 256;

  // ---- KNN graphs ----
  k_knn<<<gN, 256, 0, stream>>>(lig_x, lig_x, kNL, kNL, kLLK, 1, llidx);
  k_knn<<<gKP, 256, 0, stream>>>(kp_x, lig_x, kNK, kNL, kKLK, 0, klidx);

  // ---- encoders ----
  launch_cvt(stream, F(LIG_W), kSL + 1, kHID, 32, wEnc);
  k_build_enc<<<gN, 256, 0, stream>>>(lig_h0, tstep, kNL, kSL, 32, nA1, NLDA, kNN);
  launch_gemm(stream, nA1, NLDA, wEnc, kHID, F(LIG_B), kNN, 32, 1,
              s_tmp, kHID, nullptr, 0, nullptr, nullptr);
  k_ln<<<gN, 256, 0, stream>>>(s_tmp, nullptr, F(LIG_G), F(LIG_BETA), s_l, kNN);

  launch_cvt(stream, F(KP_W), kSK + 1, kHID, NLDA, wEnc);
  k_build_enc<<<gKP, 256, 0, stream>>>(kp_h0, tstep, kNK, kSK, NLDA, nA1, NLDA, kNKP);
  launch_gemm(stream, nA1, NLDA, wEnc, kHID, F(KP_B), kNKP, NLDA, 1,
              s_tmp, kHID, nullptr, 0, nullptr, nullptr);
  k_ln<<<gKP, 256, 0, stream>>>(s_tmp, nullptr, F(KP_G), F(KP_BETA), s_k, kNKP);

  k_zero<<<(kNN * 48 + 255) / 256, 256, 0, stream>>>(v_l, kNN * 48);

  // ---- conv layers ----
  for (int c = 0; c < kNCONV; ++c) {
    const int base = 6 + c * 50;
    // msg_ll (leaf base+20), chunked over ligand nodes
    for (int g = 0; g < 3; ++g) {
      int gb = base + 20 + g * 6;
      launch_cvt(stream, F(gb + 2), (g == 0 ? 146 : 144), kHID, ELDA, wS[g]);
      launch_cvt(stream, F(gb + 0), kHID, kVEC, kHID, wG[g]);
    }
    {
      GvpRef g1 = mkref(base + 20,      wS[0], wG[0]);
      GvpRef g2 = mkref(base + 20 + 6,  wS[1], wG[1]);
      GvpRef g3 = mkref(base + 20 + 12, wS[2], wG[2]);
      for (int n0 = 0; n0 < kNN; n0 += chunkNodes) {
        int M = chunkNodes * kLLK;
        k_build_ll<<<(M + 255) / 256, 256, 0, stream>>>(s_l, v_l, lig_x, llidx,
                                                        n0, chunkNodes, Aping, V17);
        run_msg_chain(stream, M, Aping, Apong, V17, Vu, Vg, g1, g2, g3);
        int tS = chunkNodes * kHID;
        k_agg_ll_s<<<(tS + 255) / 256, 256, 0, stream>>>(Apong, n0, chunkNodes, aggs);
        int tV = chunkNodes * 48;
        k_agg_ll_v<<<(tV + 255) / 256, 256, 0, stream>>>(Vg, n0, chunkNodes, aggv);
      }
    }
    // msg_kl (leaf base+2), chunked over kp nodes, scatter-add into agg
    for (int g = 0; g < 3; ++g) {
      int gb = base + 2 + g * 6;
      launch_cvt(stream, F(gb + 2), (g == 0 ? 146 : 144), kHID, ELDA, wS[g]);
      launch_cvt(stream, F(gb + 0), kHID, kVEC, kHID, wG[g]);
    }
    {
      GvpRef g1 = mkref(base + 2,      wS[0], wG[0]);
      GvpRef g2 = mkref(base + 2 + 6,  wS[1], wG[1]);
      GvpRef g3 = mkref(base + 2 + 12, wS[2], wG[2]);
      int kpChunk = chunkNodes * 2 > kNKP ? kNKP : chunkNodes * 2;
      for (int kp0 = 0; kp0 < kNKP; kp0 += kpChunk) {
        int nk = (kNKP - kp0 < kpChunk) ? (kNKP - kp0) : kpChunk;
        int M = nk * kKLK;
        k_build_kl<<<(M + 255) / 256, 256, 0, stream>>>(s_k, kp_v, kp_x, lig_x, klidx,
                                                        kp0, nk, Aping, V17);
        run_msg_chain(stream, M, Aping, Apong, V17, Vu, Vg, g1, g2, g3);
        k_scat_s<<<(M * kHID + 255) / 256, 256, 0, stream>>>(Apong, klidx, kp0, nk, aggs);
        k_scat_v<<<(M * 48 + 255) / 256, 256, 0, stream>>>(Vg, klidx, kp0, nk, aggv);
      }
    }
    // update GVPs (leaf base+38, base+44)
    {
      int u0 = base + 38, u1 = base + 44;
      launch_cvt(stream, F(u0 + 2), 288, kHID, NLDA, wS[0]);
      launch_cvt(stream, F(u1 + 2), 144, kHID, ELDA, wS[1]);
      launch_cvt(stream, F(u0 + 0), kHID, kVEC, kHID, wG[0]);
      launch_cvt(stream, F(u1 + 0), kHID, kVEC, kHID, wG[1]);
      k_build_upd<<<gN, 256, 0, stream>>>(s_l, aggs, v_l, aggv, nA1, nV32);
      k_gvp_vec<32,32,16><<<gN, 256, 0, stream>>>(nV32, F(u0 + 1), F(u0 + 3), kNN,
                                                  nA1, NLDA, 256, 288, nVu);
      launch_gemm(stream, nA1, NLDA, wS[0], kHID, F(u0 + 5), kNN, 288, 1,
                  nullptr, 0, nA2, NLDA, nullptr, nullptr);
      launch_gemm(stream, nA2, NLDA, wG[0], kVEC, F(u0 + 4), kNN, kHID, 2,
                  nullptr, 0, nullptr, 0, nVu, nVg);
      k_gvp_vec<16,16,16><<<gN, 256, 0, stream>>>(nVg, F(u1 + 1), F(u1 + 3), kNN,
                                                  nA2, NLDA, 128, 160, nVu);
      launch_gemm(stream, nA2, NLDA, wS[1], kHID, F(u1 + 5), kNN, 160, 1,
                  s_tmp, kHID, nA1, NLDA, nullptr, nullptr);
      launch_gemm(stream, nA1, NLDA, wG[1], kVEC, F(u1 + 4), kNN, kHID, 2,
                  nullptr, 0, nullptr, 0, nVu, nVg);
      k_ln<<<gN, 256, 0, stream>>>(s_tmp, s_l, F(base + 1), F(base + 0), s_l, kNN);
      k_add<<<(kNN * 48 + 255) / 256, 256, 0, stream>>>(v_l, nVg, kNN * 48);
    }
  }

  // ---- noise GVPs + output projection ----
  {
    int n0 = NOISE0, n1 = NOISE0 + 6, n2 = NOISE0 + 12;
    launch_cvt(stream, F(n0 + 2), 144, kHID, ELDA, wS[0]);
    launch_cvt(stream, F(n1 + 2), 144, kHID, ELDA, wS[1]);
    launch_cvt(stream, F(n2 + 2), 144, 64, ELDA, wS[2]);
    launch_cvt(stream, F(n0 + 0), kHID, kVEC, kHID, wG[0]);
    launch_cvt(stream, F(n1 + 0), kHID, kVEC, kHID, wG[1]);
    launch_cvt(stream, F(OUT_W), 64, 16, 64, wOut);

    k_castrow<<<(kNN * kHID + 255) / 256, 256, 0, stream>>>(s_l, nA1, NLDA, kHID, kNN);
    k_gvp_vec<16,16,16><<<gN, 256, 0, stream>>>(v_l, F(n0 + 1), F(n0 + 3), kNN,
                                                nA1, NLDA, 128, 160, nVu);
    launch_gemm(stream, nA1, NLDA, wS[0], kHID, F(n0 + 5), kNN, 160, 1,
                nullptr, 0, nA2, NLDA, nullptr, nullptr);
    launch_gemm(stream, nA2, NLDA, wG[0], kVEC, F(n0 + 4), kNN, kHID, 2,
                nullptr, 0, nullptr, 0, nVu, nVg);
    k_gvp_vec<16,16,16><<<gN, 256, 0, stream>>>(nVg, F(n1 + 1), F(n1 + 3), kNN,
                                                nA2, NLDA, 128, 160, nVu);
    launch_gemm(stream, nA2, NLDA, wS[1], kHID, F(n1 + 5), kNN, 160, 1,
                nullptr, 0, nA1, NLDA, nullptr, nullptr);
    launch_gemm(stream, nA1, NLDA, wG[1], kVEC, F(n1 + 4), kNN, kHID, 2,
                nullptr, 0, nullptr, 0, nVu, nVg);
    k_gvp_vec<16,16,1><<<gN, 256, 0, stream>>>(nVg, F(n2 + 1), F(n2 + 3), kNN,
                                               nA1, NLDA, 128, 160, nVu);
    launch_gemm(stream, nA1, NLDA, wS[2], 64, F(n2 + 5), kNN, 160, 1,
                s_tmp, 64, s3h, 64, nullptr, nullptr);

    float* out  = (float*)d_out;
    float* outV = out + (size_t)kNN * kSL;
    k_gate1<<<gN, 256, 0, stream>>>(s_tmp, F(n2 + 0), F(n2 + 4), nVu, outV, kNN);
    launch_gemm(stream, s3h, 64, wOut, 16, F(OUT_B), kNN, 64, 0,
                out, 16, nullptr, 0, nullptr, nullptr);
  }
}
